// LocallyConnected_10153302687851
// MI455X (gfx1250) — matrix operand order repaired and checked by reading, not compiled, not run on hardware
//
#include <hip/hip_runtime.h>

// locally-connected dims
#define H_    32
#define W_    32
#define FIN_  64
#define FOUT_ 64
#define OH_   30
#define OW_   30
#define L_    (OH_ * OW_)     // 900
#define NB_   128             // batch
#define KTOT_ 576             // 9*64
#define LDA_  72              // padded LDS row stride (ushorts); 144B, 16B aligned
#define LDB_  72

typedef __bf16 bf16_t;
typedef bf16_t v16bf __attribute__((ext_vector_type(16)));
typedef float  v8f   __attribute__((ext_vector_type(8)));

union Frag {
    v16bf v;
    uint4 q[2];
};

__device__ __forceinline__ unsigned short f2bf_rne(float x) {
    unsigned int u = __float_as_uint(x);
    unsigned int r = 0x7FFFu + ((u >> 16) & 1u);
    return (unsigned short)((u + r) >> 16);
}
__device__ __forceinline__ float bf2f(unsigned short h) {
    return __uint_as_float(((unsigned int)h) << 16);
}

// Split one float into (hi, lo) bf16 pair: x ~= hi + lo.
__device__ __forceinline__ void split1(float x, unsigned short& h, unsigned short& lo) {
    h  = f2bf_rne(x);
    lo = f2bf_rne(x - bf2f(h));
}

// Split a float4 and store 4 hi bf16 + 4 lo bf16 as one b64 each.
__device__ __forceinline__ void split_store4(unsigned short* __restrict__ hi,
                                             unsigned short* __restrict__ lo,
                                             float4 v) {
    unsigned short h0, h1, h2, h3, l0, l1, l2, l3;
    split1(v.x, h0, l0); split1(v.y, h1, l1);
    split1(v.z, h2, l2); split1(v.w, h3, l3);
    uint2 ph, pl;
    ph.x = (unsigned)h0 | ((unsigned)h1 << 16);
    ph.y = (unsigned)h2 | ((unsigned)h3 << 16);
    pl.x = (unsigned)l0 | ((unsigned)l1 << 16);
    pl.y = (unsigned)l2 | ((unsigned)l3 << 16);
    *(uint2*)hi = ph;    // ds_store_b64
    *(uint2*)lo = pl;    // ds_store_b64
}

// One block per output location l. 1024 threads = 32 waves.
// Wave w computes C tile (mtile = w>>2, ntile = w&3) of the 128x64 per-location GEMM.
__global__ __launch_bounds__(1024)
void lc_wmma_bf16x3_kernel(const float* __restrict__ X,
                           const float* __restrict__ Wf,
                           float* __restrict__ Out) {
    __shared__ __align__(16) unsigned short As_hi[NB_ * LDA_];
    __shared__ __align__(16) unsigned short As_lo[NB_ * LDA_];
    __shared__ __align__(16) unsigned short Bs_hi[FOUT_ * LDB_];
    __shared__ __align__(16) unsigned short Bs_lo[FOUT_ * LDB_];

    const int l  = blockIdx.x;        // 0..899
    const int oy = l / OW_;
    const int ox = l % OW_;

    const int tid   = threadIdx.x;
    const int lane  = tid & 31;
    const int wave  = tid >> 5;
    const int mtile = wave >> 2;      // 0..7
    const int ntile = wave & 3;       // 0..3

    // ---- staging geometry (constant across chunks) ----
    // A: thread covers rows n0 and n0+64, 4 consecutive fin at kq
    const int n0 = tid >> 4;               // 0..63
    const int kq = (tid & 15) * 4;         // 0,4,..,60
    // B: thread covers row bk, 4 consecutive fout at bf (stored transposed)
    const int bk = tid >> 4;               // 0..63
    const int bf = (tid & 15) * 4;         // 0,4,..,60
    const float* Wl = Wf + (size_t)l * (KTOT_ * FOUT_);

    // ---- fragment geometry (16-bit A 16x32 / B 32x16 lane layouts, wave32) ----
    const int half = lane >> 4;
    const int mrow = mtile * 16 + (lane & 15);
    const int kA0  = half ? 8 : 0;         // A q[0]: K kA0..kA0+7
    const int kA1  = half ? 24 : 16;       // A q[1]: K kA1..kA1+7
    const int fcol = ntile * 16 + (lane & 15);
    const int kB0  = half * 8;             // B q[0]: K kB0..+7, q[1]: +16..+23

    const unsigned short* Ah_base = &As_hi[mrow * LDA_];
    const unsigned short* Al_base = &As_lo[mrow * LDA_];
    const unsigned short* Bh_base = &Bs_hi[fcol * LDB_];
    const unsigned short* Bl_base = &Bs_lo[fcol * LDB_];

    v8f acc = {};

    // ---- software pipeline: global loads for chunk ch+1 issued before compute(ch)
    float4 ra0, ra1, rb;
    {
        const int base = (oy * W_ + ox) * FIN_;   // chunk 0: fh=0, fw=0
        ra0 = *(const float4*)(X + (size_t)n0 * (H_ * W_ * FIN_) + base + kq);
        ra1 = *(const float4*)(X + (size_t)(n0 + 64) * (H_ * W_ * FIN_) + base + kq);
        rb  = *(const float4*)(Wl + bk * FOUT_ + bf);
    }

    for (int ch = 0; ch < 9; ++ch) {
        // ---- convert staged registers -> LDS (hi/lo bf16 planes)
        split_store4(&As_hi[n0 * LDA_ + kq], &As_lo[n0 * LDA_ + kq], ra0);
        split_store4(&As_hi[(n0 + 64) * LDA_ + kq], &As_lo[(n0 + 64) * LDA_ + kq], ra1);
        {   // B transposed: value j goes to row (bf+j), column bk
            unsigned short h, lo;
            split1(rb.x, h, lo); Bs_hi[(bf + 0) * LDB_ + bk] = h; Bs_lo[(bf + 0) * LDB_ + bk] = lo;
            split1(rb.y, h, lo); Bs_hi[(bf + 1) * LDB_ + bk] = h; Bs_lo[(bf + 1) * LDB_ + bk] = lo;
            split1(rb.z, h, lo); Bs_hi[(bf + 2) * LDB_ + bk] = h; Bs_lo[(bf + 2) * LDB_ + bk] = lo;
            split1(rb.w, h, lo); Bs_hi[(bf + 3) * LDB_ + bk] = h; Bs_lo[(bf + 3) * LDB_ + bk] = lo;
        }
        __syncthreads();

        // ---- prefetch next chunk's global data (latency hidden under WMMAs)
        if (ch < 8) {
            const int nc = ch + 1;
            const int base = ((oy + nc / 3) * W_ + (ox + nc % 3)) * FIN_;
            ra0 = *(const float4*)(X + (size_t)n0 * (H_ * W_ * FIN_) + base + kq);
            ra1 = *(const float4*)(X + (size_t)(n0 + 64) * (H_ * W_ * FIN_) + base + kq);
            rb  = *(const float4*)(Wl + (nc * 64 + bk) * FOUT_ + bf);
        }

        // ---- two K=32 WMMA steps, 3 split-bf16 products each
#pragma unroll
        for (int kc = 0; kc < 2; ++kc) {
            const int kb = kc * 32;
            Frag ah, al, bh, bl;
            ah.q[0] = *(const uint4*)(Ah_base + kb + kA0);
            ah.q[1] = *(const uint4*)(Ah_base + kb + kA1);
            al.q[0] = *(const uint4*)(Al_base + kb + kA0);
            al.q[1] = *(const uint4*)(Al_base + kb + kA1);
            bh.q[0] = *(const uint4*)(Bh_base + kb + kB0);
            bh.q[1] = *(const uint4*)(Bh_base + kb + kB0 + 16);
            bl.q[0] = *(const uint4*)(Bl_base + kb + kB0);
            bl.q[1] = *(const uint4*)(Bl_base + kb + kB0 + 16);

            acc = __builtin_amdgcn_wmma_f32_16x16x32_bf16(
                false, ah.v, false, bh.v, (short)0, acc, false, false);
            acc = __builtin_amdgcn_wmma_f32_16x16x32_bf16(
                false, ah.v, false, bl.v, (short)0, acc, false, false);
            acc = __builtin_amdgcn_wmma_f32_16x16x32_bf16(
                false, al.v, false, bh.v, (short)0, acc, false, false);
        }
        __syncthreads();
    }

    // ---- store C tile: VGPR r -> row (half*8 + r), lane&15 -> column
#pragma unroll
    for (int r = 0; r < 8; ++r) {
        const int m = mtile * 16 + half * 8 + r;   // batch index
        Out[((size_t)m * L_ + l) * FOUT_ + fcol] = acc[r];
    }
}

extern "C" void kernel_launch(void* const* d_in, const int* in_sizes, int n_in,
                              void* d_out, int out_size, void* d_ws, size_t ws_size,
                              hipStream_t stream) {
    const float* X  = (const float*)d_in[0];   // [128,32,32,64] f32
    const float* Wf = (const float*)d_in[1];   // [900,576,64]   f32
    float* Out      = (float*)d_out;           // [128,30,30,64] f32
    (void)in_sizes; (void)n_in; (void)out_size; (void)d_ws; (void)ws_size;

    lc_wmma_bf16x3_kernel<<<L_, 1024, 0, stream>>>(X, Wf, Out);
}
